// MDRNN2D_75136157877050
// MI455X (gfx1250) — matrix operand order repaired and checked by reading, not compiled, not run on hardware
//
#include <hip/hip_runtime.h>
#include <hip/hip_bf16.h>

// ---------------------------------------------------------------------------
// MDRNN2D (4-direction 2D GRU) for MI455X / gfx1250
//   B=32, IN=64, H=W=64 (63x63 computed region), HID=128, gates=3*HID=384
//
// Anti-diagonal wavefront; one workgroup per (cell, direction).
// Per-cell gate block (32 x 384) via v_wmma_f32_16x16x32_bf16:
//   K-steps: 2 (x@Wx) + 4 (h_up@Wh) + 4 (h_left@Wh2); 60 WMMA per wave.
// Staging uses gfx1250 async-to-LDS copies (ASYNCcnt); x is pre-converted
// to a contiguous bf16 per-cell layout once per call; weights pre-packed
// into WMMA B-tile bf16 layout and kept L2-resident (<1 MB).
// ---------------------------------------------------------------------------

#define BSZ   32
#define INCH  64
#define HIMG  64
#define WIMG  64
#define HID   128
#define G3    384          // 3*HID
#define NT    24           // 384/16 column tiles
#define CELLN 63           // computed region is 63x63
#define HG_CELL (BSZ * HID)          // 4096 halfs per cell
#define HG_DIR  (64 * 64 * HG_CELL)  // halo-padded 64x64 grid per direction
#define XB_CELL (BSZ * INCH)         // 2048 halfs per cell of packed x

typedef __attribute__((ext_vector_type(16))) __bf16 v16bf;
typedef __attribute__((ext_vector_type(8)))  __bf16 v8bf;
typedef __attribute__((ext_vector_type(8)))  float  v8f;

__device__ __forceinline__ v8f wmma_bf16(v16bf a, v16bf b, v8f c) {
    return __builtin_amdgcn_wmma_f32_16x16x32_bf16(
        /*neg_a=*/false, a, /*neg_b=*/false, b,
        /*c_mod=*/(short)0, c, /*reuse_a=*/false, /*reuse_b=*/false);
}

// A-fragment (16x32 bf16) load per ISA layout:
// lane<16 : row m=lane,    K = k0+{0..7, 16..23}
// lane>=16: row m=lane-16, K = k0+{8..15, 24..31}
__device__ __forceinline__ v16bf load_a(const __bf16* base, int stride,
                                        int m, int k0, int lane) {
    const __bf16* p = base + m * stride + k0 + ((lane >= 16) ? 8 : 0);
    v8bf lo = *(const v8bf*)(p);
    v8bf hi = *(const v8bf*)(p + 16);
    return __builtin_shufflevector(lo, hi, 0, 1, 2, 3, 4, 5, 6, 7,
                                           8, 9, 10, 11, 12, 13, 14, 15);
}

__device__ __forceinline__ float sigmoidf_(float v) {
    return 1.0f / (1.0f + __expf(-v));
}

// One K=32 step: 3 gate-column B tiles (r,z,n) x 2 batch-row A tiles = 6 WMMA.
__device__ __forceinline__ void gemm_step(const __bf16* abase, int astride, int k0,
                                          const __bf16* bstep, int lane, int wv,
                                          v8f* c_r, v8f* c_z, v8f* c_n) {
    v16bf br = *(const v16bf*)(bstep + (size_t)(wv)      * 512 + lane * 16);
    v16bf bz = *(const v16bf*)(bstep + (size_t)(wv + 8)  * 512 + lane * 16);
    v16bf bn = *(const v16bf*)(bstep + (size_t)(wv + 16) * 512 + lane * 16);
    v16bf a0 = load_a(abase, astride, (lane & 15),      k0, lane);
    v16bf a1 = load_a(abase, astride, (lane & 15) + 16, k0, lane);
    c_r[0] = wmma_bf16(a0, br, c_r[0]);
    c_r[1] = wmma_bf16(a1, br, c_r[1]);
    c_z[0] = wmma_bf16(a0, bz, c_z[0]);
    c_z[1] = wmma_bf16(a1, bz, c_z[1]);
    c_n[0] = wmma_bf16(a0, bn, c_n[0]);
    c_n[1] = wmma_bf16(a1, bn, c_n[1]);
}

// gfx1250 async copy: 16 bytes per lane, global -> LDS, tracked by ASYNCcnt.
__device__ __forceinline__ void async_copy16(unsigned lds_off, const void* gaddr) {
    asm volatile("global_load_async_to_lds_b128 %0, %1, off"
                 :: "v"(lds_off), "v"(gaddr) : "memory");
}
__device__ __forceinline__ void wait_async0() {
    asm volatile("s_wait_asynccnt 0" ::: "memory");
}

// ---------------------------------------------------------------------------
// init: fill output with ones (uncomputed border cells must stay 1.0)
// ---------------------------------------------------------------------------
__global__ void ones_kernel(float* __restrict__ out, int n) {
    for (int i = blockIdx.x * blockDim.x + threadIdx.x; i < n;
         i += gridDim.x * blockDim.x)
        out[i] = 1.0f;
}

// init: zero the halo row 0 / col 0 of each direction's h-grid
__global__ void halo_kernel(__bf16* __restrict__ hgrid) {
    const int total = 4 * 127 * HG_CELL;
    for (int idx = blockIdx.x * blockDim.x + threadIdx.x; idx < total;
         idx += gridDim.x * blockDim.x) {
        int e = idx & (HG_CELL - 1);
        int t = idx >> 12;          // /4096
        int dd = t / 127;
        int c  = t % 127;
        int yy = (c < 64) ? 0 : (c - 63);
        int xx = (c < 64) ? c : 0;
        hgrid[((size_t)(dd * 64 + yy) * 64 + xx) * HG_CELL + e] = (__bf16)0.0f;
    }
}

// ---------------------------------------------------------------------------
// pack x (B, IN, 64, 64) f32 -> bf16 [yi][xi][b][ic] (contiguous per cell)
// ---------------------------------------------------------------------------
__global__ void pack_x_kernel(const float* __restrict__ src,
                              __bf16* __restrict__ dst) {
    const int total = CELLN * CELLN * XB_CELL;
    for (int idx = blockIdx.x * blockDim.x + threadIdx.x; idx < total;
         idx += gridDim.x * blockDim.x) {
        int ic = idx & 63;
        int b  = (idx >> 6) & 31;
        int cell = idx >> 11;
        int xi = cell % CELLN;
        int yi = cell / CELLN;
        dst[idx] = (__bf16)src[(((size_t)b * INCH + ic) * HIMG + yi) * WIMG + xi];
    }
}

// ---------------------------------------------------------------------------
// pack f32 weights (4, K, 384) -> bf16 WMMA B-tile layout [d][kt][nt][lane][16]
//   lane<16 : n = nt*16+lane,    k = kt*32 + {0..7, 16..23}[e]
//   lane>=16: n = nt*16+lane-16, k = kt*32 + {8..15, 24..31}[e]
// ---------------------------------------------------------------------------
__global__ void pack_b_kernel(const float* __restrict__ src,
                              __bf16* __restrict__ dst, int K) {
    const int KT = K / 32;
    const int total = 4 * KT * NT * 512;
    for (int idx = blockIdx.x * blockDim.x + threadIdx.x; idx < total;
         idx += gridDim.x * blockDim.x) {
        int e    = idx & 15;
        int lane = (idx >> 4) & 31;
        int nt   = (idx >> 9) % NT;
        int kt   = ((idx >> 9) / NT) % KT;
        int dd   = (idx >> 9) / (NT * KT);
        int k = kt * 32 + ((lane >= 16) ? 8 : 0) + e + ((e >= 8) ? 8 : 0);
        int n = nt * 16 + (lane & 15);
        dst[idx] = (__bf16)src[((size_t)dd * K + k) * G3 + n];
    }
}

// ---------------------------------------------------------------------------
// One anti-diagonal step. grid = (cells_on_diag, 4 directions), block = 256.
// ---------------------------------------------------------------------------
__global__ void __launch_bounds__(256)
cell_kernel(const __bf16* __restrict__ xbf, const float* __restrict__ bias,
            const __bf16* __restrict__ wxp, const __bf16* __restrict__ whp,
            const __bf16* __restrict__ wh2p, __bf16* __restrict__ hgrid,
            float* __restrict__ out, int t, int y0) {
    __shared__ __align__(16) __bf16 sh_x[XB_CELL];      // 4 KB
    __shared__ __align__(16) __bf16 sh_hup[BSZ * HID];  // 8 KB
    __shared__ __align__(16) __bf16 sh_hl[BSZ * HID];   // 8 KB

    const int d = blockIdx.y;                  // direction
    const int y = y0 + blockIdx.x;             // canonical scan coords
    const int xq = t - y;
    const int fy = d & 1;
    const int fx = (d >> 1) & 1;
    const int yi = fy ? (CELLN - 1 - y)  : y;  // image coords of this cell's x
    const int xi = fx ? (CELLN - 1 - xq) : xq;
    const int yo = yi + fy;                    // output coords
    const int xo = xi + fx;

    const int lane = threadIdx.x & 31;
    const int wv   = threadIdx.x >> 5;         // 0..7 = hidden column tile
    const int hi8  = (lane >> 4) & 1;

    // Kick off async global->LDS staging (1280 x 16B chunks, ASYNCcnt).
    {
        const char* xs = (const char*)(xbf + (size_t)(yi * CELLN + xi) * XB_CELL);
        const char* us = (const char*)(hgrid +
            ((size_t)(d * 64 + y) * 64 + (xq + 1)) * HG_CELL);      // h_up cell
        const char* ls = (const char*)(hgrid +
            ((size_t)(d * 64 + y + 1) * 64 + xq) * HG_CELL);        // h_left cell
        const unsigned off_x  = (unsigned)(size_t)&sh_x[0];
        const unsigned off_up = (unsigned)(size_t)&sh_hup[0];
        const unsigned off_l  = (unsigned)(size_t)&sh_hl[0];
#pragma unroll
        for (int i = 0; i < 5; ++i) {          // 1280 chunks / 256 threads
            int c = threadIdx.x + i * 256;
            if (c < 256)       async_copy16(off_x  + c * 16,         xs + c * 16);
            else if (c < 768)  async_copy16(off_up + (c - 256) * 16, us + (size_t)(c - 256) * 16);
            else               async_copy16(off_l  + (c - 768) * 16, ls + (size_t)(c - 768) * 16);
        }
    }

    // Overlap: warm L2/WGP$ with this direction's recurrent weights.
    {
        const __bf16* w1 = whp  + (size_t)d * 4 * NT * 512;
        const __bf16* w2 = wh2p + (size_t)d * 4 * NT * 512;
        for (int off = threadIdx.x * 64; off < 4 * NT * 512; off += 256 * 64) {
            __builtin_prefetch(w1 + off, 0, 3);
            __builtin_prefetch(w2 + off, 0, 3);
        }
    }

    wait_async0();
    __syncthreads();

    v8f c_r[2] = {};   // r-gate pre-act (bias added later)
    v8f c_z[2] = {};   // z-gate pre-act
    v8f c_xn[2] = {};  // n-gate: x@Wx part
    v8f c_hn[2] = {};  // n-gate: h@Wh + h@Wh2 part (multiplied by r)

    const __bf16* bx  = wxp  + (size_t)d * 2 * NT * 512;
    const __bf16* bh  = whp  + (size_t)d * 4 * NT * 512;
    const __bf16* bh2 = wh2p + (size_t)d * 4 * NT * 512;

#pragma unroll
    for (int ks = 0; ks < 2; ++ks)
        gemm_step(sh_x, INCH, ks * 32, bx + (size_t)ks * NT * 512,
                  lane, wv, c_r, c_z, c_xn);
#pragma unroll
    for (int ks = 0; ks < 4; ++ks)
        gemm_step(sh_hup, HID, ks * 32, bh + (size_t)ks * NT * 512,
                  lane, wv, c_r, c_z, c_hn);
#pragma unroll
    for (int ks = 0; ks < 4; ++ks)
        gemm_step(sh_hl, HID, ks * 32, bh2 + (size_t)ks * NT * 512,
                  lane, wv, c_r, c_z, c_hn);

    // Elementwise gates + h write-out.
    const int n = wv * 16 + (lane & 15);       // hidden channel
    const float br_ = bias[(size_t)d * G3 + n];
    const float bz_ = bias[(size_t)d * G3 + HID + n];
    const float bn_ = bias[(size_t)d * G3 + 2 * HID + n];
    __bf16* hcell = hgrid + ((size_t)(d * 64 + y + 1) * 64 + (xq + 1)) * HG_CELL;

#pragma unroll
    for (int rt = 0; rt < 2; ++rt) {
#pragma unroll
        for (int i = 0; i < 8; ++i) {
            int m = rt * 16 + hi8 * 8 + i;     // batch row
            float hup = (float)sh_hup[m * HID + n];
            float hlf = (float)sh_hl[m * HID + n];
            float r  = sigmoidf_(c_r[rt][i] + br_);
            float z  = sigmoidf_(c_z[rt][i] + bz_);
            float nn = tanhf(c_xn[rt][i] + bn_ + r * c_hn[rt][i]);
            float h  = (1.0f - z) * nn + z * 0.5f * (hup + hlf);
            hcell[m * HID + n] = (__bf16)h;
            // out layout: (B, HID, 4, H, W)
            out[((((size_t)m * HID + n) * 4 + d) * HIMG + yo) * WIMG + xo] = h;
        }
    }
}

// ---------------------------------------------------------------------------
// host launch
// ---------------------------------------------------------------------------
extern "C" void kernel_launch(void* const* d_in, const int* in_sizes, int n_in,
                              void* d_out, int out_size, void* d_ws, size_t ws_size,
                              hipStream_t stream) {
    const float* x    = (const float*)d_in[0];
    const float* Wx   = (const float*)d_in[1];
    const float* Wh   = (const float*)d_in[2];
    const float* Wh2  = (const float*)d_in[3];
    const float* bias = (const float*)d_in[4];
    float* out = (float*)d_out;

    // workspace carve-up (halfs):
    //   hgrid: 4 * 64 * 64 * 4096 = 67,108,864   (~134 MB)
    //   xbf  : 63 * 63 * 2048     =  8,128,512   (~16 MB)
    //   WxP  : 4 * 2 * 24 * 512   =     98,304
    //   WhP  : 4 * 4 * 24 * 512   =    196,608
    //   Wh2P : same               =    196,608   (~151 MB total)
    unsigned char* wsb = (unsigned char*)d_ws;
    __bf16* hgrid = (__bf16*)wsb;
    __bf16* xbf   = hgrid + (size_t)4 * HG_DIR;
    __bf16* wxp   = xbf + (size_t)CELLN * CELLN * XB_CELL;
    __bf16* whp   = wxp + (size_t)4 * 2 * NT * 512;
    __bf16* wh2p  = whp + (size_t)4 * 4 * NT * 512;

    ones_kernel<<<4096, 256, 0, stream>>>(out, out_size);
    halo_kernel<<<512, 256, 0, stream>>>(hgrid);
    pack_x_kernel<<<2048, 256, 0, stream>>>(x, xbf);
    pack_b_kernel<<<256, 256, 0, stream>>>(Wx,  wxp,  INCH);
    pack_b_kernel<<<256, 256, 0, stream>>>(Wh,  whp,  HID);
    pack_b_kernel<<<256, 256, 0, stream>>>(Wh2, wh2p, HID);

    // 125 anti-diagonals; all 4 directions run concurrently per step.
    for (int t = 0; t <= 2 * (CELLN - 1); ++t) {
        int y0 = (t > CELLN - 1) ? (t - (CELLN - 1)) : 0;
        int y1 = (t < CELLN - 1) ? t : (CELLN - 1);
        dim3 grid(y1 - y0 + 1, 4);
        cell_kernel<<<grid, 256, 0, stream>>>(xbf, bias, wxp, whp, wh2p,
                                              hgrid, out, t, y0);
    }
}
